// LEDDecoderAttentionTopicAware_35235911696706
// MI455X (gfx1250) — matrix operand order repaired and checked by reading, not compiled, not run on hardware
//
#include <hip/hip_runtime.h>
#include <cstdint>

// ---------------------------------------------------------------------------
// Types for CDNA5 (gfx1250) wave32 WMMA
// ---------------------------------------------------------------------------
typedef __bf16 bf16;
typedef __attribute__((ext_vector_type(16))) __bf16 v16bf;
typedef __attribute__((ext_vector_type(8)))  __bf16 bv8;   // 16 bytes
typedef __attribute__((ext_vector_type(8)))  float  v8f;
typedef __attribute__((ext_vector_type(4)))  unsigned int u32x4;
typedef __attribute__((ext_vector_type(4)))  int  i32x4;
typedef __attribute__((ext_vector_type(8)))  int  i32x8;

// Problem constants
#define BATCH 2
#define TGT   1024
#define SRC   2048
#define EMB   1024
#define TDIM  512
#define NH    16
#define HD    64
#define QSCALE 0.125f   // 64^-0.5

__device__ __forceinline__ v8f wmma_bf16(v16bf a, v16bf b, v8f c) {
  return __builtin_amdgcn_wmma_f32_16x16x32_bf16(false, a, false, b,
                                                 (short)0, c, false, false);
}

// Build a v16bf fragment from two contiguous 8-element (16B) chunks.
__device__ __forceinline__ v16bf mk_frag(const bf16* lo, const bf16* hi) {
  const bv8 a = *(const bv8*)lo;
  const bv8 b = *(const bv8*)hi;
  v16bf f;
#pragma unroll
  for (int i = 0; i < 8; ++i) { f[i] = a[i]; f[i + 8] = b[i]; }
  return f;
}

__device__ __forceinline__ v8f zero8() {
  v8f z;
#pragma unroll
  for (int i = 0; i < 8; ++i) z[i] = 0.f;
  return z;
}

// Per-lane async DMA: 16B global -> LDS, tracked by ASYNCcnt.
__device__ __forceinline__ void async_ld_b128(void* lds, const void* gptr) {
  uint32_t l = (uint32_t)(uintptr_t)lds;   // LDS aperture: low 32 bits = offset
  asm volatile("global_load_async_to_lds_b128 %0, %1, off"
               :: "v"(l), "v"(gptr) : "memory");
}
__device__ __forceinline__ void wait_async0() {
  asm volatile("s_wait_asynccnt 0x0" ::: "memory");
}

// TDM: 1-D tile of `nelem` bf16 (contiguous in memory) -> LDS, inserting
// `pad_amount` DWORDs of padding every 2^pad_interval 8-byte units.
// Descriptor per CDNA5 ISA 8.3-8.6; one call = one tensor_load_to_lds.
// (This toolchain's builtin takes 6 args: g0, g1, g2, g3, extra group, cpol.)
__device__ __forceinline__ void tdm_load_1d_bf16(void* lds, const void* gptr,
                                                 int nelem, int pad_int_code,
                                                 int pad_amt_code) {
  uint64_t ga = (uint64_t)(uintptr_t)gptr;
  u32x4 g0;
  g0[0] = 1u;                                  // count=1 (valid descriptor)
  g0[1] = (uint32_t)(uintptr_t)lds;            // lds_addr
  g0[2] = (uint32_t)ga;                        // global_addr[31:0]
  g0[3] = (uint32_t)((ga >> 32) & 0x01FFFFFFu) // global_addr[56:32]
          | (2u << 30);                        // type=2 ("image")
  i32x8 g1;
  g1[0] = (1 << 16)                            // data_size=1 -> 2 bytes
        | (1 << 20)                            // pad_enable
        | (pad_int_code << 22)                 // pad_interval
        | (pad_amt_code << 25);                // pad_amount
  g1[1] = (nelem & 0xFFFF) << 16;              // tensor_dim0[15:0] @ bits 63:48
  g1[2] = ((nelem >> 16) & 0xFFFF)             // tensor_dim0[31:16]
        | (1 << 16);                           // tensor_dim1 = 1
  g1[3] = (nelem & 0xFFFF) << 16;              // tile_dim0 @ bits 127:112
  g1[4] = 0;                                   // tile_dim1=0 (1-D), tile_dim2=0
  g1[5] = nelem;                               // tensor_dim0_stride[31:0]
  g1[6] = 0;
  g1[7] = 0;
  i32x4 z4; z4[0] = z4[1] = z4[2] = z4[3] = 0; // groups 2/3 unused (<=2D)
  i32x8 z8;
#pragma unroll
  for (int i = 0; i < 8; ++i) z8[i] = 0;
  __builtin_amdgcn_tensor_load_to_lds(g0, g1, z4, z4, z8, 0);
}

// ---------------------------------------------------------------------------
// fp32 -> bf16 conversion (4 elems / thread)
// ---------------------------------------------------------------------------
__global__ __launch_bounds__(256) void cvt_kernel(const float* __restrict__ s,
                                                  bf16* __restrict__ d, int n) {
  int i = (blockIdx.x * 256 + threadIdx.x) * 4;
  if (i + 3 < n) {
    float4 v = *(const float4*)&s[i];
    d[i + 0] = (bf16)v.x; d[i + 1] = (bf16)v.y;
    d[i + 2] = (bf16)v.z; d[i + 3] = (bf16)v.w;
  }
}

// ---------------------------------------------------------------------------
// Generic NT GEMM:  C(M,N) = A(M,K) @ W(N,K)^T + bias, bf16 in, fp32 acc.
// mode 0: fp32 row-major out.  mode 1: bf16 head-split out (B*NH, Sper, HD),
// with (acc+bias)*alpha applied.
// 128 threads = 4 waves in 2x2; block tile 128x128x32; wave tile 64x64
// (16 WMMA per wave per K-step, 8 fragment gathers). Tiles arrive in LDS via
// per-lane async DMA (ASYNCcnt).
// ---------------------------------------------------------------------------
#define BM 128
#define BN 128
#define BK 32

__global__ __launch_bounds__(128)
void gemm_bf16_nt(const bf16* __restrict__ A, const bf16* __restrict__ W,
                  const float* __restrict__ bias, int M, int N, int K,
                  float alpha, int mode, int Sper,
                  bf16* __restrict__ outB, float* __restrict__ outF) {
  __shared__ bf16 As[BM][BK + 8];
  __shared__ bf16 Bs[BN][BK + 8];

  const int tid  = threadIdx.x;
  const int lane = tid & 31, wave = tid >> 5;
  const int wy = wave >> 1, wx = wave & 1;
  const int m0 = blockIdx.y * BM;
  const int n0 = blockIdx.x * BN;

  const int mrow = lane & 15;
  const int kbA  = (lane < 16) ? 0 : 16;   // A operand K-base for this half
  const int kbB  = (lane < 16) ? 0 : 16;   // B operand K-base
  const int rAdd = (lane < 16) ? 0 : 8;    // C/D row offset for this half

  v8f acc[4][4];
#pragma unroll
  for (int i = 0; i < 4; ++i)
#pragma unroll
    for (int j = 0; j < 4; ++j) acc[i][j] = zero8();

  for (int k0 = 0; k0 < K; k0 += BK) {
    __syncthreads();
#pragma unroll
    for (int c = 0; c < 4; ++c) {
      int chunk = tid + c * 128;          // 512 chunks of 8 bf16
      int row   = chunk >> 2;
      int col8  = (chunk & 3) * 8;
      async_ld_b128(&As[row][col8], &A[(size_t)(m0 + row) * K + k0 + col8]);
      async_ld_b128(&Bs[row][col8], &W[(size_t)(n0 + row) * K + k0 + col8]);
    }
    wait_async0();
    __syncthreads();

    v16bf af[4], bfr[4];
#pragma unroll
    for (int i = 0; i < 4; ++i) {
      int ar = wy * 64 + i * 16 + mrow;
      af[i] = mk_frag(&As[ar][kbA], &As[ar][kbA + 8]);
    }
#pragma unroll
    for (int j = 0; j < 4; ++j) {
      int bn = wx * 64 + j * 16 + mrow;
      bfr[j] = mk_frag(&Bs[bn][kbB], &Bs[bn][kbB + 8]);
    }
#pragma unroll
    for (int i = 0; i < 4; ++i)
#pragma unroll
      for (int j = 0; j < 4; ++j) acc[i][j] = wmma_bf16(af[i], bfr[j], acc[i][j]);
  }

  // epilogue
#pragma unroll
  for (int i = 0; i < 4; ++i)
#pragma unroll
    for (int j = 0; j < 4; ++j) {
      int baseM = m0 + wy * 64 + i * 16;
      int n     = n0 + wx * 64 + j * 16 + mrow;
      float bvl = bias ? bias[n] : 0.f;
#pragma unroll
      for (int r = 0; r < 8; ++r) {
        int m = baseM + r + rAdd;
        float val = (acc[i][j][r] + bvl) * alpha;
        if (mode == 0) {
          outF[(size_t)m * N + n] = val;
        } else {
          int b = m / Sper, s = m % Sper;
          int h = n >> 6, d = n & 63;
          outB[(((size_t)(b * NH + h)) * Sper + s) * HD + d] = (bf16)val;
        }
      }
    }
}

// ---------------------------------------------------------------------------
// Fused dual flash attention.
// Per (b,h) and 64 query rows: online-softmax over src for S1=q k^T and
// S2=q kt^T, accumulating O1 += P1 V, O2 += P2 Vt, O3 += P2 V.
// Block = 128 threads = 4 waves, one 16-row band per wave.
// K/Kt tiles are fetched by the Tensor Data Mover (one descriptor each,
// 1-D 8KB run with pad 4 DWORDs / 32 DWORDs to match the LDS row pitch);
// V/Vt need a transpose so they go through the manual path.
// ---------------------------------------------------------------------------
#define TS 64

__global__ __launch_bounds__(128)
void dual_flash(const bf16* __restrict__ Q, const bf16* __restrict__ Kb,
                const bf16* __restrict__ Vb, const bf16* __restrict__ Ktb,
                const bf16* __restrict__ Vtb, float* __restrict__ O1,
                float* __restrict__ O2, float* __restrict__ O3) {
  __shared__ bf16 sK [TS][HD + 8];   // [src][hd]   (row pitch 144B = 128B + 16B pad)
  __shared__ bf16 sKt[TS][HD + 8];
  __shared__ bf16 sV [HD][TS + 8];   // [hd][src]   (transposed for PV B-frags)
  __shared__ bf16 sVt[HD][TS + 8];
  __shared__ bf16 sP [4][16][TS + 8];

  const int bh  = blockIdx.y;
  const int q0  = blockIdx.x * 64;
  const int tid = threadIdx.x, lane = tid & 31, wave = tid >> 5;
  const int mrow = lane & 15;
  const int kbA  = (lane < 16) ? 0 : 16;
  const int kbB  = (lane < 16) ? 0 : 16;
  const int rAdd = (lane < 16) ? 0 : 8;

  // Q band fragments (16 x 64, K split into 0..31 / 32..63), resident all kernel.
  const size_t qrow = (size_t)bh * TGT + q0 + wave * 16 + mrow;
  v16bf qf[2];
  qf[0] = mk_frag(&Q[qrow * HD + kbA],      &Q[qrow * HD + kbA + 8]);
  qf[1] = mk_frag(&Q[qrow * HD + 32 + kbA], &Q[qrow * HD + 32 + kbA + 8]);

  v8f o1[4], o2[4], o3[4];
#pragma unroll
  for (int j = 0; j < 4; ++j) { o1[j] = zero8(); o2[j] = zero8(); o3[j] = zero8(); }
  float m1[8], l1[8], m2[8], l2[8];
#pragma unroll
  for (int r = 0; r < 8; ++r) { m1[r] = -3.0e38f; m2[r] = -3.0e38f; l1[r] = 0.f; l2[r] = 0.f; }

  for (int s0 = 0; s0 < SRC; s0 += TS) {
    __syncthreads();
    const size_t base = ((size_t)bh * SRC + s0) * HD;
    // TDM DMA for K / Kt (contiguous 4096 bf16 each), issued by wave 0.
    if (wave == 0) {
      tdm_load_1d_bf16(&sK [0][0], &Kb [base], TS * HD, /*2^4*8B=128B*/4, /*4dw*/3);
      tdm_load_1d_bf16(&sKt[0][0], &Ktb[base], TS * HD, 4, 3);
    }
    // Manual transposed copy for V / Vt (all 128 threads).
#pragma unroll
    for (int c = 0; c < 4; ++c) {
      int chunk = tid + c * 128;               // 512 chunks of 8
      int row = chunk >> 3, col8 = (chunk & 7) * 8;
      bv8 v  = *(const bv8*)&Vb [base + (size_t)row * HD + col8];
      bv8 vt = *(const bv8*)&Vtb[base + (size_t)row * HD + col8];
#pragma unroll
      for (int e = 0; e < 8; ++e) { sV[col8 + e][row] = v[e]; sVt[col8 + e][row] = vt[e]; }
    }
    if (wave == 0) __builtin_amdgcn_s_wait_tensorcnt((short)0);
    __syncthreads();

    // ---- scores: S1 = q k^T,  S2 = q kt^T  (4 tiles of 16x16 each) ----
    v8f S1[4], S2[4];
#pragma unroll
    for (int j = 0; j < 4; ++j) {
      int n = j * 16 + mrow;
      v8f z = zero8();
      z = wmma_bf16(qf[0], mk_frag(&sK[n][kbB],      &sK[n][kbB + 8]),      z);
      z = wmma_bf16(qf[1], mk_frag(&sK[n][32 + kbB], &sK[n][32 + kbB + 8]), z);
      S1[j] = z;
      z = zero8();
      z = wmma_bf16(qf[0], mk_frag(&sKt[n][kbB],      &sKt[n][kbB + 8]),      z);
      z = wmma_bf16(qf[1], mk_frag(&sKt[n][32 + kbB], &sKt[n][32 + kbB + 8]), z);
      S2[j] = z;
    }

    // ---- online softmax (row stats live per lane-half, masks 1/2/4/8) ----
#pragma unroll
    for (int r = 0; r < 8; ++r) {
      float mx = fmaxf(fmaxf(S1[0][r], S1[1][r]), fmaxf(S1[2][r], S1[3][r]));
#pragma unroll
      for (int off = 1; off < 16; off <<= 1) mx = fmaxf(mx, __shfl_xor(mx, off, 32));
      float mn = fmaxf(m1[r], mx);
      float sc = __expf(m1[r] - mn);
      m1[r] = mn;
      float rs = 0.f;
#pragma unroll
      for (int j = 0; j < 4; ++j) { float p = __expf(S1[j][r] - mn); S1[j][r] = p; rs += p; }
#pragma unroll
      for (int off = 1; off < 16; off <<= 1) rs += __shfl_xor(rs, off, 32);
      l1[r] = l1[r] * sc + rs;
#pragma unroll
      for (int j = 0; j < 4; ++j) o1[j][r] *= sc;

      float mx2 = fmaxf(fmaxf(S2[0][r], S2[1][r]), fmaxf(S2[2][r], S2[3][r]));
#pragma unroll
      for (int off = 1; off < 16; off <<= 1) mx2 = fmaxf(mx2, __shfl_xor(mx2, off, 32));
      float mn2 = fmaxf(m2[r], mx2);
      float sc2 = __expf(m2[r] - mn2);
      m2[r] = mn2;
      float rs2 = 0.f;
#pragma unroll
      for (int j = 0; j < 4; ++j) { float p = __expf(S2[j][r] - mn2); S2[j][r] = p; rs2 += p; }
#pragma unroll
      for (int off = 1; off < 16; off <<= 1) rs2 += __shfl_xor(rs2, off, 32);
      l2[r] = l2[r] * sc2 + rs2;
#pragma unroll
      for (int j = 0; j < 4; ++j) { o2[j][r] *= sc2; o3[j][r] *= sc2; }
    }

    // ---- P1 @ V : C-layout -> A-layout via per-wave LDS scratch ----
#pragma unroll
    for (int j = 0; j < 4; ++j)
#pragma unroll
      for (int r = 0; r < 8; ++r) sP[wave][r + rAdd][j * 16 + mrow] = (bf16)S1[j][r];
    asm volatile("s_wait_dscnt 0x0" ::: "memory");
    __builtin_amdgcn_wave_barrier();
    {
      v16bf p0 = mk_frag(&sP[wave][mrow][kbA],      &sP[wave][mrow][kbA + 8]);
      v16bf p1 = mk_frag(&sP[wave][mrow][32 + kbA], &sP[wave][mrow][32 + kbA + 8]);
#pragma unroll
      for (int j = 0; j < 4; ++j) {
        int n = j * 16 + mrow;
        o1[j] = wmma_bf16(p0, mk_frag(&sV[n][kbB],      &sV[n][kbB + 8]),      o1[j]);
        o1[j] = wmma_bf16(p1, mk_frag(&sV[n][32 + kbB], &sV[n][32 + kbB + 8]), o1[j]);
      }
    }
    __builtin_amdgcn_wave_barrier();

    // ---- P2 @ Vt and P2 @ V ----
#pragma unroll
    for (int j = 0; j < 4; ++j)
#pragma unroll
      for (int r = 0; r < 8; ++r) sP[wave][r + rAdd][j * 16 + mrow] = (bf16)S2[j][r];
    asm volatile("s_wait_dscnt 0x0" ::: "memory");
    __builtin_amdgcn_wave_barrier();
    {
      v16bf p0 = mk_frag(&sP[wave][mrow][kbA],      &sP[wave][mrow][kbA + 8]);
      v16bf p1 = mk_frag(&sP[wave][mrow][32 + kbA], &sP[wave][mrow][32 + kbA + 8]);
#pragma unroll
      for (int j = 0; j < 4; ++j) {
        int n = j * 16 + mrow;
        o2[j] = wmma_bf16(p0, mk_frag(&sVt[n][kbB],      &sVt[n][kbB + 8]),      o2[j]);
        o2[j] = wmma_bf16(p1, mk_frag(&sVt[n][32 + kbB], &sVt[n][32 + kbB + 8]), o2[j]);
        o3[j] = wmma_bf16(p0, mk_frag(&sV[n][kbB],       &sV[n][kbB + 8]),       o3[j]);
        o3[j] = wmma_bf16(p1, mk_frag(&sV[n][32 + kbB],  &sV[n][32 + kbB + 8]),  o3[j]);
      }
    }
    __builtin_amdgcn_wave_barrier();
  }

  // ---- finalize ----
#pragma unroll
  for (int j = 0; j < 4; ++j)
#pragma unroll
    for (int r = 0; r < 8; ++r) {
      int row = q0 + wave * 16 + r + rAdd;
      int col = j * 16 + mrow;
      size_t idx = ((size_t)bh * TGT + row) * HD + col;
      float inv1 = 1.f / l1[r], inv2 = 1.f / l2[r];
      O1[idx] = o1[j][r] * inv1;
      O2[idx] = o2[j][r] * inv2;
      O3[idx] = o3[j][r] * inv2;
    }
}

// ---------------------------------------------------------------------------
// Gating: topic_p[b,t,:] = sigmoid(p_vec[b,t,:] @ Wg^T + bg) with the
// reference's transpose/view scramble inverted per element.
// One block per (b,t); gate stored as (B*NH, TGT).
// ---------------------------------------------------------------------------
__global__ __launch_bounds__(256)
void gate_kernel(const float* __restrict__ O1, const float* __restrict__ O2,
                 const bf16* __restrict__ Qb, const float* __restrict__ Wg,
                 const float* __restrict__ bg, float* __restrict__ gate) {
  __shared__ float red[NH][256];
  const int bt = blockIdx.x;
  const int b = bt >> 10, t = bt & 1023;
  float partial[NH];
#pragma unroll
  for (int n = 0; n < NH; ++n) partial[n] = 0.f;

  for (int j = threadIdx.x; j < 3 * EMB; j += 256) {
    int g   = t * (3 * EMB) + j;         // flat index in scrambled view
    int h   = g / (3 * HD * TGT);        // 196608 per head
    int rem = g % (3 * HD * TGT);
    int c   = rem >> 10;                 // /TGT
    int tp  = rem & 1023;
    size_t base = ((size_t)(b * NH + h) * TGT + tp) * HD;
    float val;
    if (c < HD)           val = O1[base + c];
    else if (c < 2 * HD)  val = O2[base + c - HD];
    else                  val = (float)Qb[base + c - 2 * HD];
#pragma unroll
    for (int n = 0; n < NH; ++n) partial[n] += val * Wg[n * 3 * EMB + j];
  }
#pragma unroll
  for (int n = 0; n < NH; ++n) red[n][threadIdx.x] = partial[n];
  __syncthreads();
  for (int s = 128; s > 0; s >>= 1) {
    if (threadIdx.x < (unsigned)s)
#pragma unroll
      for (int n = 0; n < NH; ++n) red[n][threadIdx.x] += red[n][threadIdx.x + s];
    __syncthreads();
  }
  if (threadIdx.x < NH) {
    float sv = red[threadIdx.x][0] + bg[threadIdx.x];
    gate[((size_t)(b * NH + threadIdx.x)) * TGT + t] = 1.f / (1.f + __expf(-sv));
  }
}

// ---------------------------------------------------------------------------
// Combine: out_head = g*O1 + (1-g)*O3, un-head-split into (B*TGT, EMB) bf16.
// ---------------------------------------------------------------------------
__global__ __launch_bounds__(256)
void combine_kernel(const float* __restrict__ O1, const float* __restrict__ O3,
                    const float* __restrict__ gate, bf16* __restrict__ X) {
  size_t i = (size_t)blockIdx.x * 256 + threadIdx.x;   // B*NH*TGT*HD
  int d = i & 63;
  size_t r = i >> 6;
  int t = r & 1023;
  size_t bh = r >> 10;
  int h = bh & 15, b = (int)(bh >> 4);
  float g = gate[bh * TGT + t];
  float val = g * O1[i] + (1.f - g) * O3[i];
  X[((size_t)(b * TGT + t)) * EMB + h * HD + d] = (bf16)val;
}

// ---------------------------------------------------------------------------
// Host launcher
// ---------------------------------------------------------------------------
extern "C" void kernel_launch(void* const* d_in, const int* in_sizes, int n_in,
                              void* d_out, int out_size, void* d_ws, size_t ws_size,
                              hipStream_t stream) {
  (void)in_sizes; (void)n_in; (void)out_size; (void)ws_size;
  const float* hid = (const float*)d_in[0];
  const float* kv  = (const float*)d_in[1];
  const float* kvt = (const float*)d_in[2];
  const float* Wq  = (const float*)d_in[3];  const float* bq  = (const float*)d_in[4];
  const float* Wk  = (const float*)d_in[5];  const float* bk  = (const float*)d_in[6];
  const float* Wv  = (const float*)d_in[7];  const float* bv  = (const float*)d_in[8];
  const float* Wtk = (const float*)d_in[9];  const float* btk = (const float*)d_in[10];
  const float* Wtv = (const float*)d_in[11]; const float* btv = (const float*)d_in[12];
  const float* Wg  = (const float*)d_in[13]; const float* bg  = (const float*)d_in[14];
  const float* Wo  = (const float*)d_in[15]; const float* bo  = (const float*)d_in[16];

  char* ws = (char*)d_ws;
  size_t off = 0;
  auto alloc = [&](size_t bytes) -> char* {
    char* p = ws + off;
    off += (bytes + 255) & ~(size_t)255;
    return p;
  };
  const size_t NE_hid = (size_t)BATCH * TGT * EMB;     // 2M
  const size_t NE_kv  = (size_t)BATCH * SRC * EMB;     // 4M
  const size_t NE_kvt = (size_t)BATCH * SRC * TDIM;    // 2M
  const size_t NE_h   = (size_t)BATCH * NH * TGT * HD; // 2M (per-head q/O)
  const size_t NE_hk  = (size_t)BATCH * NH * SRC * HD; // 4M (per-head k/v)

  bf16* hidb = (bf16*)alloc(NE_hid * 2);
  bf16* kvb  = (bf16*)alloc(NE_kv * 2);
  bf16* kvtb = (bf16*)alloc(NE_kvt * 2);
  bf16* Wqb  = (bf16*)alloc((size_t)EMB * EMB * 2);
  bf16* Wkb  = (bf16*)alloc((size_t)EMB * EMB * 2);
  bf16* Wvb  = (bf16*)alloc((size_t)EMB * EMB * 2);
  bf16* Wob  = (bf16*)alloc((size_t)EMB * EMB * 2);
  bf16* Wtkb = (bf16*)alloc((size_t)EMB * TDIM * 2);
  bf16* Wtvb = (bf16*)alloc((size_t)EMB * TDIM * 2);
  bf16* qb   = (bf16*)alloc(NE_h * 2);
  bf16* kb   = (bf16*)alloc(NE_hk * 2);
  bf16* vb   = (bf16*)alloc(NE_hk * 2);
  bf16* ktb  = (bf16*)alloc(NE_hk * 2);
  bf16* vtb  = (bf16*)alloc(NE_hk * 2);
  float* O1  = (float*)alloc(NE_h * 4);
  float* O2  = (float*)alloc(NE_h * 4);
  float* O3  = (float*)alloc(NE_h * 4);
  float* gate = (float*)alloc((size_t)BATCH * NH * TGT * 4);
  bf16* Xb   = (bf16*)alloc(NE_hid * 2);

  auto cvt = [&](const float* s, bf16* d, size_t n) {
    cvt_kernel<<<(unsigned)(n / 1024), 256, 0, stream>>>(s, d, (int)n);
  };
  cvt(hid, hidb, NE_hid);
  cvt(kv, kvb, NE_kv);
  cvt(kvt, kvtb, NE_kvt);
  cvt(Wq, Wqb, (size_t)EMB * EMB);
  cvt(Wk, Wkb, (size_t)EMB * EMB);
  cvt(Wv, Wvb, (size_t)EMB * EMB);
  cvt(Wo, Wob, (size_t)EMB * EMB);
  cvt(Wtk, Wtkb, (size_t)EMB * TDIM);
  cvt(Wtv, Wtvb, (size_t)EMB * TDIM);

  // Projections (head-split bf16 outputs)
  {
    dim3 blk(128);
    // q: (2048,1024) x (1024,1024)^T, alpha=SCALE
    gemm_bf16_nt<<<dim3(EMB / BN, (BATCH * TGT) / BM), blk, 0, stream>>>(
        hidb, Wqb, bq, BATCH * TGT, EMB, EMB, QSCALE, 1, TGT, qb, nullptr);
    // k, v: (4096,1024) x (1024,1024)^T
    gemm_bf16_nt<<<dim3(EMB / BN, (BATCH * SRC) / BM), blk, 0, stream>>>(
        kvb, Wkb, bk, BATCH * SRC, EMB, EMB, 1.f, 1, SRC, kb, nullptr);
    gemm_bf16_nt<<<dim3(EMB / BN, (BATCH * SRC) / BM), blk, 0, stream>>>(
        kvb, Wvb, bv, BATCH * SRC, EMB, EMB, 1.f, 1, SRC, vb, nullptr);
    // kt, vt: (4096,512) x (1024,512)^T
    gemm_bf16_nt<<<dim3(EMB / BN, (BATCH * SRC) / BM), blk, 0, stream>>>(
        kvtb, Wtkb, btk, BATCH * SRC, EMB, TDIM, 1.f, 1, SRC, ktb, nullptr);
    gemm_bf16_nt<<<dim3(EMB / BN, (BATCH * SRC) / BM), blk, 0, stream>>>(
        kvtb, Wtvb, btv, BATCH * SRC, EMB, TDIM, 1.f, 1, SRC, vtb, nullptr);
  }

  // Fused dual flash attention
  dual_flash<<<dim3(TGT / 64, BATCH * NH), 128, 0, stream>>>(qb, kb, vb, ktb, vtb,
                                                             O1, O2, O3);

  // Gating MLP (with scrambled p_vec gather) + combine
  gate_kernel<<<BATCH * TGT, 256, 0, stream>>>(O1, O2, qb, Wg, bg, gate);
  combine_kernel<<<(unsigned)(NE_h / 256), 256, 0, stream>>>(O1, O3, gate, Xb);

  // Output projection -> fp32 d_out
  gemm_bf16_nt<<<dim3(EMB / BN, (BATCH * TGT) / BM), 128, 0, stream>>>(
      Xb, Wob, bo, BATCH * TGT, EMB, EMB, 1.f, 0, TGT, nullptr, (float*)d_out);
}
